// PeriodicSetTransformer_51273319579857
// MI455X (gfx1250) — matrix operand order repaired and checked by reading, not compiled, not run on hardware
//
#include <hip/hip_runtime.h>
#include <hip/hip_bf16.h>
#include <math.h>

// Problem constants (from the reference)
#define B_    64
#define N_    512
#define FEA_  102
#define E_    10
#define d_    256
#define H_    4
#define D_    1024
#define DH_   256      // D/H
#define L_    3
#define NDIST_ 100
#define MTOK_ (B_*N_)  // 32768

typedef unsigned short u16;
typedef __attribute__((ext_vector_type(8)))  u16    u16x8;
typedef __attribute__((ext_vector_type(16))) __bf16 bf16x16;
typedef __attribute__((ext_vector_type(8)))  float  f32x8;

union Frag { bf16x16 v; u16x8 h[2]; };

__device__ __forceinline__ u16 f2bf(float f) {
  union { float f; unsigned u; } x; x.f = f;
  unsigned r = x.u + 0x7fffu + ((x.u >> 16) & 1u);
  return (u16)(r >> 16);
}

// ---------------------------------------------------------------------------
// Weight convert + transpose: in f32 [K,Nc] -> out bf16 [Nc,K]
// ---------------------------------------------------------------------------
__global__ void convT_kernel(const float* __restrict__ in, u16* __restrict__ out,
                             int K, int Nc) {
  const int t = blockIdx.x * blockDim.x + threadIdx.x;
  if (t >= K * Nc) return;
  const int kk = t / Nc, c = t - kk * Nc;
  out[(size_t)c * K + kk] = f2bf(in[t]);
}

// ---------------------------------------------------------------------------
// WMMA bf16 GEMM: out = act[M,K] @ W[K,Nc] + bias
//   act: bf16 row-major, pitch K.  wt: bf16 [Nc, wLd] (W columns contiguous).
//   Outputs (each optional): f32 [M,Nc], bf16 [M,Nc], bf16 transposed [Nc, outTld].
//   Batched via blockIdx.z with element strides actSB/wSB/outSB.
//   Tiles: 256 thr (8 waves), 128x128 C tile, BK=32, wave -> 32x64 (2x4 WMMA).
//   Double-buffered LDS with register staging: global loads for tile t+1 are
//   issued before the WMMA work for tile t, one barrier per K-step.
// ---------------------------------------------------------------------------
__global__ __launch_bounds__(256) void gemm_bf16_kernel(
    const u16* __restrict__ act, long actSB,
    const u16* __restrict__ wt,  long wSB, int wLd,
    const float* __restrict__ bias,
    float* __restrict__ outF, u16* __restrict__ outH, u16* __restrict__ outT,
    long outSB, int outTld, int M, int K, int Nc)
{
  __shared__ __align__(16) u16 As[2][128][32];
  __shared__ __align__(16) u16 Bs[2][128][32];

  const int bz = blockIdx.z;
  act += (size_t)bz * actSB;
  wt  += (size_t)bz * wSB;
  const size_t outOff = (size_t)bz * outSB;

  const int mBase = blockIdx.y * 128;
  const int cBase = blockIdx.x * 128;
  const int tid  = threadIdx.x;
  const int lane = tid & 31;
  const int wid  = tid >> 5;      // 0..7
  const int wy = wid >> 1;        // 0..3: 32-row slab
  const int wx = wid & 1;         // 0..1: 64-col slab
  const int lr = lane & 15, lh = lane >> 4;

  f32x8 acc[2][4] = {};

  const int ldRow = tid >> 1;          // 0..127
  const int ldSeg = (tid & 1) * 16;    // 0 or 16

  const u16* gA = act + (size_t)(mBase + ldRow) * K + ldSeg;
  const u16* gW = wt  + (size_t)(cBase + ldRow) * wLd + ldSeg;

  // Prologue: stage tile 0 into LDS buffer 0.
  u16x8 ra0 = *(const u16x8*)(gA);
  u16x8 ra1 = *(const u16x8*)(gA + 8);
  u16x8 rb0 = *(const u16x8*)(gW);
  u16x8 rb1 = *(const u16x8*)(gW + 8);
  *(u16x8*)&As[0][ldRow][ldSeg]     = ra0;
  *(u16x8*)&As[0][ldRow][ldSeg + 8] = ra1;
  *(u16x8*)&Bs[0][ldRow][ldSeg]     = rb0;
  *(u16x8*)&Bs[0][ldRow][ldSeg + 8] = rb1;
  __syncthreads();

  const int nt = K >> 5;
  for (int t = 0; t < nt; ++t) {
    const int cur = t & 1;
    const bool more = (t + 1 < nt);
    if (more) {                        // issue next tile's global loads early
      const int k0 = (t + 1) << 5;
      ra0 = *(const u16x8*)(gA + k0);
      ra1 = *(const u16x8*)(gA + k0 + 8);
      rb0 = *(const u16x8*)(gW + k0);
      rb1 = *(const u16x8*)(gW + k0 + 8);
    }

    Frag a[2], b[4];
#pragma unroll
    for (int i = 0; i < 2; ++i) {      // A frag: lane = row, K halves by lane-group
      const int row = wy * 32 + i * 16 + lr;
      a[i].h[0] = *(const u16x8*)&As[cur][row][lh * 8];
      a[i].h[1] = *(const u16x8*)&As[cur][row][16 + lh * 8];
    }
#pragma unroll
    for (int j = 0; j < 4; ++j) {      // B frag: lane = column, K halves by lane-group
      const int col = wx * 64 + j * 16 + lr;
      b[j].h[0] = *(const u16x8*)&Bs[cur][col][lh * 8];
      b[j].h[1] = *(const u16x8*)&Bs[cur][col][16 + lh * 8];
    }
#pragma unroll
    for (int i = 0; i < 2; ++i)
#pragma unroll
      for (int j = 0; j < 4; ++j)
        acc[i][j] = __builtin_amdgcn_wmma_f32_16x16x32_bf16(
            false, a[i].v, false, b[j].v, (short)0, acc[i][j], false, false);

    if (more) {                        // store staged tile into the other buffer
      const int nxt = 1 - cur;
      *(u16x8*)&As[nxt][ldRow][ldSeg]     = ra0;
      *(u16x8*)&As[nxt][ldRow][ldSeg + 8] = ra1;
      *(u16x8*)&Bs[nxt][ldRow][ldSeg]     = rb0;
      *(u16x8*)&Bs[nxt][ldRow][ldSeg + 8] = rb1;
    }
    __syncthreads();
  }

#pragma unroll
  for (int i = 0; i < 2; ++i)
#pragma unroll
    for (int j = 0; j < 4; ++j) {
      const int colG = cBase + wx * 64 + j * 16 + lr;
      const float bv = bias ? bias[colG] : 0.f;
#pragma unroll
      for (int r = 0; r < 8; ++r) {    // C layout: VGPR r, lane-hi selects M+8
        const int rowG = mBase + wy * 32 + i * 16 + lh * 8 + r;
        const float v = acc[i][j][r] + bv;
        const size_t o = outOff + (size_t)rowG * Nc + colG;
        if (outF) outF[o] = v;
        if (outH) outH[o] = f2bf(v);
        if (outT) outT[(size_t)colG * outTld + rowG] = f2bf(v);
      }
    }
}

// ---------------------------------------------------------------------------
// Fused attention: per (b, 32 query rows): for each head compute q.k^T via WMMA,
// masked softmax, head-mean; then weight by w_k, renormalize rows, emit A (bf16).
// Dynamic LDS: sc[32*512] + aacc[32*512] + red[256] + rowv[32] floats.
// ---------------------------------------------------------------------------
__global__ __launch_bounds__(256) void attn_kernel(
    const u16* __restrict__ q, const u16* __restrict__ k,
    const float* __restrict__ maskv, const float* __restrict__ features,
    u16* __restrict__ A)
{
  extern __shared__ float smem[];
  float* sc   = smem;                 // 32*512
  float* aacc = smem + 32 * 512;      // 32*512
  float* red  = aacc + 32 * 512;      // 32*8
  float* rowv = red + 32 * 8;         // 32

  const int b   = blockIdx.y;
  const int q0  = blockIdx.x * 32;
  const int tid = threadIdx.x;
  const int lane = tid & 31, wid = tid >> 5;
  const int lr = lane & 15, lh = lane >> 4;
  const float scale = 0.0625f;        // 1/sqrt(256)

  for (int h = 0; h < H_; ++h) {
    // --- scores: 2 row-tiles x 32 col-tiles of 16x16, 8 tiles per wave ---
#pragma unroll
    for (int ct4 = 0; ct4 < 4; ++ct4) {
      const int ct = wid + ct4 * 8;
#pragma unroll
      for (int rt = 0; rt < 2; ++rt) {
        f32x8 acc = {};
        const u16* qp0 = q + (((size_t)b * N_ + q0 + rt * 16 + lr) * D_ + h * DH_);
        const u16* kp0 = k + (((size_t)b * N_ + ct * 16 + lr) * D_ + h * DH_);
#pragma unroll
        for (int kd = 0; kd < DH_; kd += 32) {
          Frag a, bb;
          a.h[0]  = *(const u16x8*)(qp0 + kd + lh * 8);
          a.h[1]  = *(const u16x8*)(qp0 + kd + 16 + lh * 8);
          bb.h[0] = *(const u16x8*)(kp0 + kd + lh * 8);
          bb.h[1] = *(const u16x8*)(kp0 + kd + 16 + lh * 8);
          acc = __builtin_amdgcn_wmma_f32_16x16x32_bf16(
              false, a.v, false, bb.v, (short)0, acc, false, false);
        }
#pragma unroll
        for (int r = 0; r < 8; ++r) {
          const int row = rt * 16 + lh * 8 + r;
          const int col = ct * 16 + lr;
          sc[row * N_ + col] = acc[r] * scale + maskv[b * N_ + col];
        }
      }
    }
    __syncthreads();

    // --- per-row softmax (8 threads per row, 64 cols each) ---
    {
      const int row = tid >> 3, sub = tid & 7;
      const int c0 = sub * 64;
      float mx = -3.0e38f;
      for (int c = 0; c < 64; ++c) mx = fmaxf(mx, sc[row * N_ + c0 + c]);
      red[row * 8 + sub] = mx; __syncthreads();
      if (sub == 0) { float m = red[row * 8];
        for (int j2 = 1; j2 < 8; ++j2) m = fmaxf(m, red[row * 8 + j2]);
        rowv[row] = m; }
      __syncthreads();
      const float m = rowv[row];
      float s = 0.f;
      for (int c = 0; c < 64; ++c) s += __expf(sc[row * N_ + c0 + c] - m);
      red[row * 8 + sub] = s; __syncthreads();
      if (sub == 0) { float t = 0.f;
        for (int j2 = 0; j2 < 8; ++j2) t += red[row * 8 + j2];
        rowv[row] = t; }
      __syncthreads();
      const float inv = 0.25f / rowv[row];   // mean over H=4 heads
      for (int c = 0; c < 64; ++c) {
        const float p = __expf(sc[row * N_ + c0 + c] - m) * inv;
        if (h == 0) aacc[row * N_ + c0 + c] = p;
        else        aacc[row * N_ + c0 + c] += p;
      }
    }
    __syncthreads();
  }

  // --- weight by w_k, renormalize, store bf16 ---
  const int row = tid >> 3, sub = tid & 7;
  const int c0 = sub * 64;
  float s = 0.f;
  for (int c = 0; c < 64; ++c) {
    const float wv = features[((size_t)b * N_ + c0 + c) * FEA_];
    const float aw = aacc[row * N_ + c0 + c] * wv;
    sc[row * N_ + c0 + c] = aw;
    s += aw;
  }
  red[row * 8 + sub] = s; __syncthreads();
  if (sub == 0) { float t = 0.f;
    for (int j2 = 0; j2 < 8; ++j2) t += red[row * 8 + j2];
    rowv[row] = t; }
  __syncthreads();
  const float inv = 1.f / rowv[row];
  u16* Ar = A + ((size_t)b * N_ + q0 + row) * N_;
  for (int c = 0; c < 64; ++c) Ar[c0 + c] = f2bf(sc[row * N_ + c0 + c] * inv);
}

// ---------------------------------------------------------------------------
// Embedding: x = LN(atom_fea[idx]@comp_W + comp_b + de@emb_W + emb_b)
// One block per token, 256 threads = d channels.
// ---------------------------------------------------------------------------
__global__ __launch_bounds__(256) void embed_kernel(
    const float* __restrict__ features, const float* __restrict__ atom_fea,
    const float* __restrict__ comp_W, const float* __restrict__ comp_b,
    const float* __restrict__ emb_W, const float* __restrict__ emb_b,
    const float* __restrict__ g, const float* __restrict__ bta,
    float* __restrict__ xinitF, u16* __restrict__ xH)
{
  __shared__ float sdist[NDIST_];
  __shared__ float s1[256], s2[256];
  const size_t tok = blockIdx.x;
  const int tid = threadIdx.x;
  const float* f = features + tok * FEA_;
  if (tid < NDIST_) sdist[tid] = f[1 + tid];
  const int idx = (int)f[FEA_ - 1];
  __syncthreads();

  float acc = comp_b[tid];
  const float* af = atom_fea + (size_t)idx * 92;
  for (int j = 0; j < 92; ++j) acc += af[j] * comp_W[j * d_ + tid];

  float acc2 = emb_b[tid];
  for (int j = 0; j < NDIST_; ++j) {
    const float dj = sdist[j];
    const float* ew = emb_W + (size_t)j * E_ * d_ + tid;
#pragma unroll
    for (int sE = 0; sE < E_; ++sE) {
      const float de = 1.f - (dj - (float)sE * 0.1f);
      acc2 += de * de * ew[sE * d_];
    }
  }
  const float x = acc + acc2;
  s1[tid] = x; s2[tid] = x * x; __syncthreads();
  for (int o = 128; o > 0; o >>= 1) {
    if (tid < o) { s1[tid] += s1[tid + o]; s2[tid] += s2[tid + o]; }
    __syncthreads();
  }
  const float mean = s1[0] / d_;
  const float var  = s2[0] / d_ - mean * mean;
  const float rstd = rsqrtf(var + 1e-5f);
  const float xo = (x - mean) * rstd * g[tid] + bta[tid];
  xinitF[tok * d_ + tid] = xo;
  xH[tok * d_ + tid] = f2bf(xo);
}

// mask_k = -1e9 if sum|e| == 0 else 0
__global__ __launch_bounds__(256) void mask_kernel(const float* __restrict__ e,
                                                   float* __restrict__ maskv) {
  __shared__ float s1[256];
  const size_t tok = blockIdx.x;
  const int tid = threadIdx.x;
  float s = 0.f;
  for (int i = 0; i < 4; ++i) s += fabsf(e[tok * D_ + tid + i * 256]);
  s1[tid] = s; __syncthreads();
  for (int o = 128; o > 0; o >>= 1) { if (tid < o) s1[tid] += s1[tid + o]; __syncthreads(); }
  if (tid == 0) maskv[tok] = (s1[0] == 0.f) ? -1.0e9f : 0.f;
}

// xe = LN(e + softplus(att)) -> bf16
__global__ __launch_bounds__(256) void splus_ln_kernel(
    const float* __restrict__ e, const float* __restrict__ att,
    const float* __restrict__ g, const float* __restrict__ bta,
    u16* __restrict__ xe)
{
  __shared__ float s1[256], s2[256];
  const size_t tok = blockIdx.x;
  const int tid = threadIdx.x;
  float t[4]; float s = 0.f, ss = 0.f;
#pragma unroll
  for (int i = 0; i < 4; ++i) {
    const int c = tid + i * 256;
    const float a = att[tok * D_ + c];
    const float sp = (a > 20.f) ? a : log1pf(__expf(a));
    const float v = e[tok * D_ + c] + sp;
    t[i] = v; s += v; ss += v * v;
  }
  s1[tid] = s; s2[tid] = ss; __syncthreads();
  for (int o = 128; o > 0; o >>= 1) {
    if (tid < o) { s1[tid] += s1[tid + o]; s2[tid] += s2[tid + o]; }
    __syncthreads();
  }
  const float mean = s1[0] / D_;
  const float var  = s2[0] / D_ - mean * mean;
  const float rstd = rsqrtf(var + 1e-5f);
#pragma unroll
  for (int i = 0; i < 4; ++i) {
    const int c = tid + i * 256;
    xe[tok * D_ + c] = f2bf((t[i] - mean) * rstd * g[c] + bta[c]);
  }
}

// pooled = LN(sum_n w*(x+x_init)); out = pooled @ out_W + out_b
__global__ __launch_bounds__(256) void pool_out_kernel(
    const float* __restrict__ features, const float* __restrict__ xcur,
    const float* __restrict__ xinit, const float* __restrict__ g,
    const float* __restrict__ bta, const float* __restrict__ outW,
    const float* __restrict__ outb, float* __restrict__ out)
{
  __shared__ float s1[256], s2[256];
  const int b = blockIdx.x;
  const int c = threadIdx.x;
  float acc = 0.f;
  for (int n = 0; n < N_; ++n) {
    const size_t tok = (size_t)b * N_ + n;
    acc += features[tok * FEA_] * (xcur[tok * d_ + c] + xinit[tok * d_ + c]);
  }
  s1[c] = acc; s2[c] = acc * acc; __syncthreads();
  for (int o = 128; o > 0; o >>= 1) {
    if (c < o) { s1[c] += s1[c + o]; s2[c] += s2[c + o]; }
    __syncthreads();
  }
  const float mean = s1[0] / d_;
  const float var  = s2[0] / d_ - mean * mean;
  const float rstd = rsqrtf(var + 1e-5f);
  const float p = (acc - mean) * rstd * g[c] + bta[c];
  __syncthreads();
  s1[c] = p * outW[c]; __syncthreads();
  for (int o = 128; o > 0; o >>= 1) { if (c < o) s1[c] += s1[c + o]; __syncthreads(); }
  if (c == 0) out[b] = s1[0] + outb[0];
}

// ---------------------------------------------------------------------------
extern "C" void kernel_launch(void* const* d_in, const int* in_sizes, int n_in,
                              void* d_out, int out_size, void* d_ws, size_t ws_size,
                              hipStream_t stream) {
  (void)in_sizes; (void)n_in; (void)out_size; (void)ws_size;
  const float* features  = (const float*)d_in[0];
  const float* atom_fea  = (const float*)d_in[1];
  const float* comp_W    = (const float*)d_in[2];
  const float* comp_b    = (const float*)d_in[3];
  const float* emb_W     = (const float*)d_in[4];
  const float* emb_b     = (const float*)d_in[5];
  const float* ln_g      = (const float*)d_in[6];
  const float* ln_b      = (const float*)d_in[7];
  const float* enc_emb_W = (const float*)d_in[8];
  const float* enc_emb_b = (const float*)d_in[9];
  const float* enc_Wq_W  = (const float*)d_in[10];
  const float* enc_Wq_b  = (const float*)d_in[11];
  const float* enc_Wk_W  = (const float*)d_in[12];
  const float* enc_Wk_b  = (const float*)d_in[13];
  const float* enc_Wv_W  = (const float*)d_in[14];
  const float* enc_Wv_b  = (const float*)d_in[15];
  const float* enc_inq_W = (const float*)d_in[16];
  const float* enc_inq_b = (const float*)d_in[17];
  const float* enc_ink_W = (const float*)d_in[18];
  const float* enc_ink_b = (const float*)d_in[19];
  const float* enc_ln_g  = (const float*)d_in[20];
  const float* enc_ln_b  = (const float*)d_in[21];
  const float* enc_out_W = (const float*)d_in[22];
  const float* enc_out_b = (const float*)d_in[23];
  const float* out_W     = (const float*)d_in[24];
  const float* out_b     = (const float*)d_in[25];
  float* out = (float*)d_out;

  // Workspace carving (256B aligned)
  char* base = (char*)d_ws;
  size_t off = 0;
  auto alloc = [&](size_t bytes) -> void* {
    void* r = base + off;
    off = (off + bytes + 255) & ~(size_t)255;
    return r;
  };
  u16*   embWt  = (u16*)alloc((size_t)L_ * D_ * d_ * 2);
  u16*   WqT    = (u16*)alloc((size_t)L_ * D_ * D_ * 2);
  u16*   WkT    = (u16*)alloc((size_t)L_ * D_ * D_ * 2);
  u16*   WvT    = (u16*)alloc((size_t)L_ * D_ * D_ * 2);
  u16*   inqT   = (u16*)alloc((size_t)L_ * D_ * D_ * 2);
  u16*   inkT   = (u16*)alloc((size_t)L_ * D_ * D_ * 2);
  u16*   outWt  = (u16*)alloc((size_t)L_ * d_ * D_ * 2);
  float* xinitF = (float*)alloc((size_t)MTOK_ * d_ * 4);
  float* xcurF  = (float*)alloc((size_t)MTOK_ * d_ * 4);
  u16*   xH     = (u16*)alloc((size_t)MTOK_ * d_ * 2);
  float* eF     = (float*)alloc((size_t)MTOK_ * D_ * 4);
  u16*   eH     = (u16*)alloc((size_t)MTOK_ * D_ * 2);
  u16*   tmpH   = (u16*)alloc((size_t)MTOK_ * D_ * 2);
  u16*   qH     = (u16*)alloc((size_t)MTOK_ * D_ * 2);
  u16*   kH     = (u16*)alloc((size_t)MTOK_ * D_ * 2);
  u16*   vtH    = (u16*)alloc((size_t)D_ * MTOK_ * 2);   // [D][B*N] transposed
  u16*   AH     = (u16*)alloc((size_t)B_ * N_ * N_ * 2);
  float* attF   = (float*)alloc((size_t)MTOK_ * D_ * 4);
  float* maskv  = (float*)alloc((size_t)MTOK_ * 4);

  auto convT = [&](const float* in, u16* o, int K, int Nc) {
    const int tot = K * Nc;
    convT_kernel<<<(tot + 255) / 256, 256, 0, stream>>>(in, o, K, Nc);
  };
  auto gemm = [&](const u16* act, long actSB, const u16* wt, long wSB, int wLd,
                  const float* bias, float* oF, u16* oH, u16* oT,
                  long oSB, int oTld, int M, int K, int Nc, int nb) {
    dim3 grd(Nc / 128, M / 128, nb);
    gemm_bf16_kernel<<<grd, 256, 0, stream>>>(act, actSB, wt, wSB, wLd, bias,
                                              oF, oH, oT, oSB, oTld, M, K, Nc);
  };

  // 1) Weight conversion (bf16, transposed to [Nc,K])
  for (int l = 0; l < L_; ++l) {
    convT(enc_emb_W + (size_t)l * d_ * D_, embWt + (size_t)l * D_ * d_, d_, D_);
    convT(enc_Wq_W  + (size_t)l * D_ * D_, WqT  + (size_t)l * D_ * D_, D_, D_);
    convT(enc_Wk_W  + (size_t)l * D_ * D_, WkT  + (size_t)l * D_ * D_, D_, D_);
    convT(enc_Wv_W  + (size_t)l * D_ * D_, WvT  + (size_t)l * D_ * D_, D_, D_);
    convT(enc_inq_W + (size_t)l * D_ * D_, inqT + (size_t)l * D_ * D_, D_, D_);
    convT(enc_ink_W + (size_t)l * D_ * D_, inkT + (size_t)l * D_ * D_, D_, D_);
    convT(enc_out_W + (size_t)l * D_ * d_, outWt + (size_t)l * d_ * D_, D_, d_);
  }

  // 2) Embedding + LN -> x_init (f32) and x (bf16)
  embed_kernel<<<MTOK_, 256, 0, stream>>>(features, atom_fea, comp_W, comp_b,
                                          emb_W, emb_b, ln_g, ln_b, xinitF, xH);

  const size_t attnSmem = (size_t)(32 * 512 * 2 + 32 * 8 + 32) * sizeof(float);

  // 3) Encoder layers
  for (int l = 0; l < L_; ++l) {
    // e = x @ embW + b  (f32 for residual/mask, bf16 for chained GEMMs)
    gemm(xH, 0, embWt + (size_t)l * D_ * d_, 0, d_, enc_emb_b + (size_t)l * D_,
         eF, eH, nullptr, 0, 0, MTOK_, d_, D_, 1);
    mask_kernel<<<MTOK_, 256, 0, stream>>>(eF, maskv);

    // q = (e@Wq + b) @ inq + b
    gemm(eH, 0, WqT + (size_t)l * D_ * D_, 0, D_, enc_Wq_b + (size_t)l * D_,
         nullptr, tmpH, nullptr, 0, 0, MTOK_, D_, D_, 1);
    gemm(tmpH, 0, inqT + (size_t)l * D_ * D_, 0, D_, enc_inq_b + (size_t)l * D_,
         nullptr, qH, nullptr, 0, 0, MTOK_, D_, D_, 1);
    // k = (e@Wk + b) @ ink + b
    gemm(eH, 0, WkT + (size_t)l * D_ * D_, 0, D_, enc_Wk_b + (size_t)l * D_,
         nullptr, tmpH, nullptr, 0, 0, MTOK_, D_, D_, 1);
    gemm(tmpH, 0, inkT + (size_t)l * D_ * D_, 0, D_, enc_ink_b + (size_t)l * D_,
         nullptr, kH, nullptr, 0, 0, MTOK_, D_, D_, 1);
    // v = e@Wv + b, stored transposed [D][B*N] for the batched A@v GEMM
    gemm(eH, 0, WvT + (size_t)l * D_ * D_, 0, D_, enc_Wv_b + (size_t)l * D_,
         nullptr, nullptr, vtH, 0, MTOK_, MTOK_, D_, D_, 1);

    // A = normalize(w * mean_h softmax(qk^T/sqrt(dh) + mask))
    attn_kernel<<<dim3(N_ / 32, B_), 256, attnSmem, stream>>>(qH, kH, maskv,
                                                              features, AH);

    // att = A @ v   (batched over B; v^T rows are contiguous per batch)
    gemm(AH, (long)N_ * N_, vtH, (long)N_, MTOK_, nullptr,
         attF, nullptr, nullptr, (long)N_ * D_, 0, N_, N_, D_, B_);

    // xe = LN(e + softplus(att)) -> bf16 (reuse tmpH)
    splus_ln_kernel<<<MTOK_, 256, 0, stream>>>(eF, attF,
                                               enc_ln_g + (size_t)l * D_,
                                               enc_ln_b + (size_t)l * D_, tmpH);
    // x = xe @ outW + b  (f32 for pooling, bf16 for next layer)
    gemm(tmpH, 0, outWt + (size_t)l * d_ * D_, 0, D_, enc_out_b + (size_t)l * d_,
         xcurF, xH, nullptr, 0, 0, MTOK_, D_, d_, 1);
  }

  // 4) pooled = LN(sum_n w*(x + x_init)); out = pooled @ out_W + out_b
  pool_out_kernel<<<B_, 256, 0, stream>>>(features, xcurF, xinitF, ln_g, ln_b,
                                          out_W, out_b, out);
}
